// CubeMoveHead_43971875176948
// MI455X (gfx1250) — matrix operand order repaired and checked by reading, not compiled, not run on hardware
//
#include <hip/hip_runtime.h>
#include <hip/hip_bf16.h>

#define NNODES 500000
#define DDIM   128
#define GDIM   128
#define BSZ    512
#define MCUBES 64
#define MMOVES 24
#define HDIM   128
#define KDIM   256
#define CHUNK  2048
#define NEGV  (-1000000000.0f)

// dynamic LDS layout (bytes)
#define W1S_OFF  0        // 128*256 bf16 = 65536
#define W2S_OFF  65536    // 32*128  bf16 = 8192   (contiguous after w1s, matches ws layout)
#define HB_OFF   73728    // 8 waves * 16*128 bf16 = 32768
#define SMEM_SZ  106496

typedef __attribute__((ext_vector_type(16))) __bf16 v16bf;
typedef __attribute__((ext_vector_type(8)))  float  v8f;
typedef unsigned int u32x4 __attribute__((ext_vector_type(4)));
typedef int          i32x8 __attribute__((ext_vector_type(8)));
typedef int          i32x4 __attribute__((ext_vector_type(4)));

union Frag { uint4 q[2]; v16bf v; };

__device__ __forceinline__ void put4(v16bf& a, int o, float4 f) {
    a[o + 0] = (__bf16)f.x; a[o + 1] = (__bf16)f.y;
    a[o + 2] = (__bf16)f.z; a[o + 3] = (__bf16)f.w;
}

// ---------------- weight pre-conversion: fp32 -> transposed bf16 ----------------
__global__ __launch_bounds__(256)
void convert_kernel(const float* __restrict__ W1, const float* __restrict__ W2,
                    __bf16* __restrict__ w1t, __bf16* __restrict__ w2t) {
    int idx = blockIdx.x * 256 + threadIdx.x;
    if (idx < HDIM * KDIM) {                       // W1t[n][k] = W1[k][n]
        int n = idx >> 8, k = idx & 255;
        w1t[n * KDIM + k] = (__bf16)W1[k * HDIM + n];
    }
    int i2 = idx - HDIM * KDIM;
    if (i2 >= 0 && i2 < 32 * HDIM) {               // W2t[m][k] = W2[k][m] (pad m>=24 with 0)
        int m = i2 >> 7, k = i2 & 127;
        w2t[m * HDIM + k] = (m < MMOVES) ? (__bf16)W2[k * MMOVES + m] : (__bf16)0.0f;
    }
}

// ---------------- init output to NEG, zero per-batch counters ----------------
__global__ __launch_bounds__(256)
void init_kernel(float* __restrict__ out, int n, int* __restrict__ counts) {
    int i = blockIdx.x * 256 + threadIdx.x;
    if (i < n) out[i] = NEGV;
    if (i < BSZ) counts[i] = 0;
}

// ---------------- pass A: chunk sums of cube_mask + per-batch counts ----------------
__global__ __launch_bounds__(256)
void count_kernel(const unsigned char* __restrict__ cm, const int* __restrict__ batch,
                  int* __restrict__ chunkSums, int* __restrict__ counts) {
    __shared__ int lc[BSZ];
    __shared__ int red[256];
    int tid = threadIdx.x;
    for (int i = tid; i < BSZ; i += 256) lc[i] = 0;
    __syncthreads();
    int base = blockIdx.x * CHUNK + tid * 8;
    int s = 0;
    for (int j = 0; j < 8; ++j) {
        int i = base + j;
        if (i < NNODES) {
            int m = cm[i] ? 1 : 0;
            s += m;
            if (m) atomicAdd(&lc[batch[i]], 1);
        }
    }
    red[tid] = s;
    __syncthreads();
    for (int off = 128; off > 0; off >>= 1) {
        if (tid < off) red[tid] += red[tid + off];
        __syncthreads();
    }
    if (tid == 0) chunkSums[blockIdx.x] = red[0];
    for (int i = tid; i < BSZ; i += 256) {
        int v = lc[i];
        if (v) atomicAdd(&counts[i], v);
    }
}

// ---------------- pass B: single-block scans ----------------
__global__ __launch_bounds__(512)
void scan_kernel(const int* __restrict__ counts, int* __restrict__ offsets,
                 const int* __restrict__ chunkSums, int* __restrict__ chunkOff, int nchunks) {
    __shared__ int s[BSZ];
    int tid = threadIdx.x;
    int own = counts[tid];
    s[tid] = own;
    for (int off = 1; off < BSZ; off <<= 1) {
        __syncthreads();
        int v = (tid >= off) ? s[tid - off] : 0;
        __syncthreads();
        s[tid] += v;
    }
    offsets[tid] = s[tid] - own;   // exclusive per-batch cube offset
    if (tid == 0) {
        int run = 0;
        for (int i = 0; i < nchunks; ++i) { chunkOff[i] = run; run += chunkSums[i]; }
    }
}

// ---------------- pass C: per-node cube index (-1 if not a scattered row) ----------------
__global__ __launch_bounds__(256)
void cidx_kernel(const unsigned char* __restrict__ cm, const int* __restrict__ batch,
                 const int* __restrict__ offsets, const int* __restrict__ chunkOff,
                 int* __restrict__ cidx) {
    __shared__ int s[256];
    int tid = threadIdx.x;
    int base = blockIdx.x * CHUNK + tid * 8;
    int m[8]; int sum = 0;
    for (int j = 0; j < 8; ++j) {
        int i = base + j;
        m[j] = (i < NNODES && cm[i]) ? 1 : 0;
        sum += m[j];
    }
    int own = sum;
    s[tid] = sum;
    for (int off = 1; off < 256; off <<= 1) {
        __syncthreads();
        int v = (tid >= off) ? s[tid - off] : 0;
        __syncthreads();
        s[tid] += v;
    }
    int run = chunkOff[blockIdx.x] + (s[tid] - own);  // exclusive prefix for this thread
    for (int j = 0; j < 8; ++j) {
        int i = base + j;
        if (i < NNODES) {
            run += m[j];                               // inclusive cumsum P[i]
            int ci = run - 1 - offsets[batch[i]];
            cidx[i] = (m[j] && ci >= 0 && ci < MCUBES) ? ci : -1;
        }
    }
}

// ---------------- main fused WMMA MLP + scatter ----------------
__global__ __launch_bounds__(256)
void gemm_scatter(const float* __restrict__ nf, const float* __restrict__ gf,
                  const float* __restrict__ b1, const float* __restrict__ b2,
                  const __bf16* __restrict__ w1t, const __bf16* __restrict__ w2t,
                  const int* __restrict__ batch, const int* __restrict__ cidx,
                  const unsigned char* __restrict__ mv, float* __restrict__ out) {
    extern __shared__ char smem[];
    __bf16* w1s = (__bf16*)(smem + W1S_OFF);        // [128][256] B operand, layer 1
    __bf16* w2s = (__bf16*)(smem + W2S_OFF);        // [32][128]  B operand, layer 2
    const int lane = threadIdx.x & 31;
    const int w    = threadIdx.x >> 5;
    const int half = lane >> 4;                     // 0: lanes 0-15, 1: lanes 16-31
    const int l16  = lane & 15;
    __bf16* hb = (__bf16*)(smem + HB_OFF) + w * (16 * HDIM);  // per-wave 16x128 h tile
    const int rowBase = blockIdx.x * 128 + w * 16;

    // -------- stage both weight tables (72 KB, contiguous) into LDS via TDM --------
#if defined(__gfx1250__) && __has_builtin(__builtin_amdgcn_tensor_load_to_lds)
    if (w == 0) {
        // Tensor DMA: 2D tile, 144 rows x 256 bf16 elements (= w1t 64KB + w2t 8KB)
        unsigned long long ga = (unsigned long long)(uintptr_t)w1t;   // tile start in global
        unsigned ldsOff = (unsigned)(uintptr_t)(void*)smem;           // LDS byte offset
        u32x4 g0;
        g0[0] = 1u;                                           // count=1, user mode, no gather
        g0[1] = ldsOff;                                       // lds_addr
        g0[2] = (unsigned)(ga & 0xffffffffu);                 // global_addr[31:0]
        g0[3] = (unsigned)((ga >> 32) & 0x1ffffffu) | (2u << 30);  // global_addr[56:32] | type=2
        i32x8 g1;
        g1[0] = 0x00010000;                                   // workgroup_mask=0, data_size=1 (2B)
        g1[1] = (int)(256u << 16);                            // tensor_dim0 = 256
        g1[2] = (int)(144u << 16);                            // tensor_dim1 = 144
        g1[3] = (int)(256u << 16);                            // tile_dim0   = 256
        g1[4] = 144;                                          // tile_dim1   = 144, tile_dim2=0
        g1[5] = 256;                                          // tensor_dim0_stride = 256
        g1[6] = 0;
        g1[7] = 0;
        i32x4 g2 = {0, 0, 0, 0};
        i32x4 g3 = {0, 0, 0, 0};
        i32x8 gx = {0, 0, 0, 0, 0, 0, 0, 0};                  // extra group (clang-23 form), zero
        __builtin_amdgcn_tensor_load_to_lds(g0, g1, g2, g3, gx, 0);
        __builtin_amdgcn_s_wait_tensorcnt(0);
    }
#else
    {
        const uint4* s1 = (const uint4*)w1t;  uint4* d1 = (uint4*)w1s;
        #pragma unroll
        for (int i = 0; i < 16; ++i) d1[threadIdx.x + 256 * i] = s1[threadIdx.x + 256 * i];
        const uint4* s2 = (const uint4*)w2t;  uint4* d2 = (uint4*)w2s;
        #pragma unroll
        for (int i = 0; i < 2; ++i)  d2[threadIdx.x + 256 * i] = s2[threadIdx.x + 256 * i];
    }
#endif
    __syncthreads();

    int r = rowBase + l16; if (r >= NNODES) r = NNODES - 1;
    const float* nrow = nf + (long)r * DDIM;
    const float* grow = gf + (long)batch[r] * GDIM;

    // -------- A fragments (16x32 bf16) for the 8 K-tiles of layer 1 --------
    // lanes 0-15 hold K={0..7,16..23}; lanes 16-31 hold K={8..15,24..31}
    v16bf afr[8];
    #pragma unroll
    for (int kt = 0; kt < 8; ++kt) {
        const float* p = (kt < 4) ? (nrow + kt * 32) : (grow + (kt - 4) * 32);
        const int ko = half * 8;
        float4 r0 = *(const float4*)(p + ko);
        float4 r1 = *(const float4*)(p + ko + 4);
        float4 r2 = *(const float4*)(p + 16 + ko);
        float4 r3 = *(const float4*)(p + 16 + ko + 4);
        v16bf a;
        put4(a, 0, r0); put4(a, 4, r1); put4(a, 8, r2); put4(a, 12, r3);
        afr[kt] = a;
    }

    // -------- layer 1: zero-init acc, round-buffered B frags, 64 WMMAs --------
    v8f acc[8];
    #pragma unroll
    for (int nt = 0; nt < 8; ++nt)
        #pragma unroll
        for (int i = 0; i < 8; ++i) acc[nt][i] = 0.0f;

    const __bf16* bbase = w1s + l16 * KDIM + half * 8;    // + nt*16*KDIM + kt*32
    auto loadB1 = [&](int kt, int nt) -> Frag {
        const __bf16* p = bbase + nt * (16 * KDIM) + kt * 32;
        Frag f;
        f.q[0] = *(const uint4*)p;
        f.q[1] = *(const uint4*)(p + 16);
        return f;
    };

    Frag bc[8];
    #pragma unroll
    for (int nt = 0; nt < 8; ++nt) bc[nt] = loadB1(0, nt);
    #pragma unroll
    for (int kt = 0; kt < 8; ++kt) {
        Frag bn[8];                                        // prefetch whole next round
        #pragma unroll
        for (int nt = 0; nt < 8; ++nt) bn[nt] = loadB1(kt < 7 ? kt + 1 : 7, nt);
        #pragma unroll
        for (int nt = 0; nt < 8; ++nt)
            acc[nt] = __builtin_amdgcn_wmma_f32_16x16x32_bf16(
                false, afr[kt], false, bc[nt].v, (short)0, acc[nt], false, false);
        #pragma unroll
        for (int nt = 0; nt < 8; ++nt) bc[nt] = bn[nt];
    }

    // -------- bias + ReLU, relayout C->A via per-wave LDS --------
    #pragma unroll
    for (int nt = 0; nt < 8; ++nt) {
        int col = nt * 16 + l16;
        float bv = b1[col];
        #pragma unroll
        for (int v = 0; v < 8; ++v) {
            float x = acc[nt][v] + bv;
            x = x > 0.0f ? x : 0.0f;
            hb[(half * 8 + v) * HDIM + col] = (__bf16)x;  // C layout: M = v + 8*half, N = col
        }
    }

    // -------- layer 2: h[16x128] @ W2t, 2 N-tiles x 4 K-tiles --------
    v8f acc2[2];
    #pragma unroll
    for (int nt = 0; nt < 2; ++nt)
        #pragma unroll
        for (int i = 0; i < 8; ++i) acc2[nt][i] = 0.0f;
    #pragma unroll
    for (int kt = 0; kt < 4; ++kt) {
        const int k0 = kt * 32 + half * 8;
        Frag a2;
        a2.q[0] = *(const uint4*)(hb + l16 * HDIM + k0);        // K = k0..k0+7
        a2.q[1] = *(const uint4*)(hb + l16 * HDIM + k0 + 16);   // K = k0+16..k0+23
        #pragma unroll
        for (int nt = 0; nt < 2; ++nt) {
            const __bf16* bp = w2s + (nt * 16 + l16) * HDIM + kt * 32 + half * 8;
            Frag f;
            f.q[0] = *(const uint4*)bp;
            f.q[1] = *(const uint4*)(bp + 16);
            acc2[nt] = __builtin_amdgcn_wmma_f32_16x16x32_bf16(
                false, a2.v, false, f.v, (short)0, acc2[nt], false, false);
        }
    }

    // -------- masked scatter into logits (bias fused here) --------
    #pragma unroll
    for (int nt = 0; nt < 2; ++nt) {
        int n = nt * 16 + l16;
        if (n < MMOVES) {
            float bv = b2[n];
            #pragma unroll
            for (int v = 0; v < 8; ++v) {
                int rr = rowBase + half * 8 + v;
                if (rr < NNODES) {
                    int ci = cidx[rr];
                    if (ci >= 0) {
                        long o = ((long)batch[rr] * MCUBES + ci) * MMOVES + n;
                        if (mv[o]) out[o] = acc2[nt][v] + bv;
                    }
                }
            }
        }
    }
}

extern "C" void kernel_launch(void* const* d_in, const int* in_sizes, int n_in,
                              void* d_out, int out_size, void* d_ws, size_t ws_size,
                              hipStream_t stream) {
    const float* nf = (const float*)d_in[0];          // node_features [N,128]
    const float* gf = (const float*)d_in[1];          // global_features [512,128]
    const float* W1 = (const float*)d_in[2];          // [256,128]
    const float* b1 = (const float*)d_in[3];          // [128]
    const float* W2 = (const float*)d_in[4];          // [128,24]
    const float* b2 = (const float*)d_in[5];          // [24]
    const unsigned char* cm = (const unsigned char*)d_in[6];   // cube_mask bool [N]
    const int*   batch = (const int*)d_in[7];         // [N] sorted
    const unsigned char* mv = (const unsigned char*)d_in[8];   // move_mask bool [512,64,24]
    float* out = (float*)d_out;

    char* ws = (char*)d_ws;
    __bf16* w1t      = (__bf16*)(ws + 0);             // 64 KB
    __bf16* w2t      = (__bf16*)(ws + 65536);         // 8 KB (contiguous with w1t)
    int*    cidx     = (int*)(ws + 73728);            // 4*N = 2,000,000 B
    int*    counts   = (int*)(ws + 2073728);          // 2 KB
    int*    offsets  = (int*)(ws + 2075776);          // 2 KB
    int*    chunkSum = (int*)(ws + 2077824);          // 1 KB
    int*    chunkOff = (int*)(ws + 2078848);          // 1 KB

    const int nchunks = (NNODES + CHUNK - 1) / CHUNK; // 245

    convert_kernel<<<(HDIM * KDIM + 32 * HDIM + 255) / 256, 256, 0, stream>>>(W1, W2, w1t, w2t);
    init_kernel<<<(out_size + 255) / 256, 256, 0, stream>>>(out, out_size, counts);
    count_kernel<<<nchunks, 256, 0, stream>>>(cm, batch, chunkSum, counts);
    scan_kernel<<<1, 512, 0, stream>>>(counts, offsets, chunkSum, chunkOff, nchunks);
    cidx_kernel<<<nchunks, 256, 0, stream>>>(cm, batch, offsets, chunkOff, cidx);
    gemm_scatter<<<(NNODES + 127) / 128, 256, SMEM_SZ, stream>>>(nf, gf, b1, b2, w1t, w2t,
                                                                 batch, cidx, mv, out);
}
